// PairwiseAttentionHead_88940182765944
// MI455X (gfx1250) — compile-verified, matrix operand order checked by reading, not run on hardware
//
#include <hip/hip_runtime.h>
#include <hip/hip_bf16.h>

// Problem constants (from reference)
#define B_   4
#define N_   8192
#define D_   128
#define E_   262144
#define KTOT 256               // 2*D (concat src||dst)
#define LDSW 264               // 256 + 8 bf16 pad -> 16B pad per row, kills bank conflicts

typedef __attribute__((ext_vector_type(16))) __bf16 v16bf;
typedef __attribute__((ext_vector_type(8)))  __bf16 v8bf;
typedef __attribute__((ext_vector_type(8)))  float  v8f;
typedef __attribute__((ext_vector_type(4)))  float  v4f;
typedef __attribute__((ext_vector_type(4)))  unsigned int v4u;

// ---------------------------------------------------------------------------
// Kernel 1: zero the 1 GiB output with wide nontemporal stores (memory floor).
// ---------------------------------------------------------------------------
__global__ __launch_bounds__(256) void zero_kernel(v4f* __restrict__ out, long n4) {
    long i      = (long)blockIdx.x * blockDim.x + threadIdx.x;
    long stride = (long)gridDim.x * blockDim.x;
    v4f z = {0.0f, 0.0f, 0.0f, 0.0f};
    for (; i < n4; i += stride) {
        __builtin_nontemporal_store(z, &out[i]);
    }
}

// ---------------------------------------------------------------------------
// Kernel 2: per wave, one 16-edge tile x one batch:
//   hid = silu([h_src||h_dst] (16x256, bf16) @ W1 (256x128, bf16 in LDS))
//   score = hid @ W2 + b2 ; scatter into out[b, src, dst]
//
// Loop structure: accumulators for all 8 N-tiles stay resident (64 VGPRs);
// A and B fragments are transient (8 VGPRs each). __syncthreads() at the top
// of the tile loop fences LDS so ds_loads are NOT hoisted+spilled to scratch.
// ---------------------------------------------------------------------------
__global__ __launch_bounds__(256) void edge_score_kernel(
    const float* __restrict__ h,
    const long long* __restrict__ ei,      // edge_index, int64, [2, E]
    const float* __restrict__ W1,          // [256,128] row-major
    const float* __restrict__ b1,          // [128]
    const float* __restrict__ W2,          // [128,1]
    const float* __restrict__ b2p,         // [1]
    float* __restrict__ out,               // [B,N,N]
    int nWaves)
{
    __shared__ alignas(16) unsigned short w1t[D_ * LDSW];   // W1^T as bf16, padded rows

    // ---- Stage W1^T into LDS (coalesced global reads, one-time per block) ----
    for (int idx = threadIdx.x; idx < KTOT * D_; idx += blockDim.x) {
        int k = idx >> 7;          // row of W1 (K index)
        int n = idx & 127;         // col of W1 (hidden index)
        __bf16 v = (__bf16)W1[idx];
        w1t[n * LDSW + k] = __builtin_bit_cast(unsigned short, v);
    }

    const int lane = threadIdx.x & 31;
    const int m    = lane & 15;        // edge row within tile
    const int s    = lane >> 4;        // lane half (selects K sub-chunks)
    const int wave = blockIdx.x * (blockDim.x >> 5) + (threadIdx.x >> 5);
    const float b2 = b2p[0];

    // Per-lane hidden-column scalars, invariant across tiles: hoist once.
    float biasr[8], w2r[8];
#pragma unroll
    for (int nt = 0; nt < 8; ++nt) {
        biasr[nt] = b1[nt * 16 + m];
        w2r[nt]   = W2[nt * 16 + m];
    }

    const int nUnits = (E_ / 16) * B_;   // 65536; with 8192 waves -> exactly 8 iters/wave
    for (int unit = wave; unit < nUnits; unit += nWaves) {
        // LDS fence: uniform trip count across all waves in the block; prevents
        // the compiler from hoisting the 64 B-fragment ds_loads out of the loop.
        __syncthreads();

        const int b    = unit & 3;
        const int tile = unit >> 2;
        const int e    = tile * 16 + m;

        const long long si = ei[e];
        const long long di = ei[E_ + e];
        const float* ps = h + ((long)b * N_ + si) * D_;
        const float* pd = h + ((long)b * N_ + di) * D_;

        v8f acc[8] = {};    // C tiles for all 8 N-tiles (resident)

#pragma unroll
        for (int kt = 0; kt < 8; ++kt) {
            // ---- Gather A fragment: one 16x32 bf16 slice, ISA A-layout ----
            // lanes 0-15: VGPR0..3 = K {0..7}, VGPR4..7 = K {16..23}
            // lanes16-31: VGPR0..3 = K {8..15}, VGPR4..7 = K {24..31}
            const float* base = (kt < 4 ? ps : pd) + (kt & 3) * 32;
            v4f c0a = *(const v4f*)(base + s * 8);
            v4f c0b = *(const v4f*)(base + s * 8 + 4);
            v4f c1a = *(const v4f*)(base + 16 + s * 8);
            v4f c1b = *(const v4f*)(base + 16 + s * 8 + 4);
            v16bf a;
            a[0]  = (__bf16)c0a.x; a[1]  = (__bf16)c0a.y; a[2]  = (__bf16)c0a.z; a[3]  = (__bf16)c0a.w;
            a[4]  = (__bf16)c0b.x; a[5]  = (__bf16)c0b.y; a[6]  = (__bf16)c0b.z; a[7]  = (__bf16)c0b.w;
            a[8]  = (__bf16)c1a.x; a[9]  = (__bf16)c1a.y; a[10] = (__bf16)c1a.z; a[11] = (__bf16)c1a.w;
            a[12] = (__bf16)c1b.x; a[13] = (__bf16)c1b.y; a[14] = (__bf16)c1b.z; a[15] = (__bf16)c1b.w;

#pragma unroll
            for (int nt = 0; nt < 8; ++nt) {
                const int n = nt * 16 + m;           // hidden column owned by this lane
                // B fragment: column n of W1, K chunks matching A layout; 2x ds b128
                const v4u q0 = *(const v4u*)&w1t[n * LDSW + kt * 32 + s * 8];
                const v4u q1 = *(const v4u*)&w1t[n * LDSW + kt * 32 + 16 + s * 8];
                v8bf lo = __builtin_bit_cast(v8bf, q0);
                v8bf hi = __builtin_bit_cast(v8bf, q1);
                v16bf bf = __builtin_shufflevector(lo, hi,
                    0, 1, 2, 3, 4, 5, 6, 7, 8, 9, 10, 11, 12, 13, 14, 15);
                acc[nt] = __builtin_amdgcn_wmma_f32_16x16x32_bf16(
                    false, a, false, bf, (short)0, acc[nt], false, false);
            }
        }

        // ---- epilogue: bias + silu (fast rcp) + dot with W2 ----
        float part[8] = {};
#pragma unroll
        for (int nt = 0; nt < 8; ++nt) {
            const float bias = biasr[nt];
            const float w2v  = w2r[nt];
#pragma unroll
            for (int r = 0; r < 8; ++r) {
                float x  = acc[nt][r] + bias;
                float sg = __builtin_amdgcn_rcpf(1.0f + __expf(-x));   // fast sigmoid
                part[r] += x * sg * w2v;
            }
        }

        // ---- reduce across the 16-lane half (wave32 xor shuffles) ----
#pragma unroll
        for (int r = 0; r < 8; ++r) {
            float v = part[r];
            v += __shfl_xor(v, 1, 32);
            v += __shfl_xor(v, 2, 32);
            v += __shfl_xor(v, 4, 32);
            v += __shfl_xor(v, 8, 32);
            part[r] = v;
        }

        // ---- scatter 16 scores: lane0 -> rows 0..7, lane16 -> rows 8..15 ----
        if (m == 0) {
#pragma unroll
            for (int r = 0; r < 8; ++r) {
                const int mm = s * 8 + r;
                const int ee = tile * 16 + mm;
                const long long ssi = ei[ee];
                const long long ddi = ei[E_ + ee];
                out[((long)b * N_ + ssi) * N_ + ddi] = part[r] + b2;
            }
        }
    }
}

// ---------------------------------------------------------------------------
extern "C" void kernel_launch(void* const* d_in, const int* in_sizes, int n_in,
                              void* d_out, int out_size, void* d_ws, size_t ws_size,
                              hipStream_t stream) {
    (void)in_sizes; (void)n_in; (void)out_size; (void)d_ws; (void)ws_size;

    const float*     h   = (const float*)d_in[0];
    const long long* ei  = (const long long*)d_in[1];
    const float*     W1  = (const float*)d_in[2];
    const float*     b1  = (const float*)d_in[3];
    const float*     W2  = (const float*)d_in[4];
    const float*     b2  = (const float*)d_in[5];
    float*           out = (float*)d_out;

    // 1) zero the B*N*N output (1 GiB) — this is the roofline floor
    const long n4 = (long)B_ * N_ * N_ / 4;
    zero_kernel<<<8192, 256, 0, stream>>>((v4f*)out, n4);

    // 2) WMMA edge-score compute + scatter
    const int blocks = 1024;                  // 8 waves each -> 8192 waves
    const int nWaves = blocks * 8;
    edge_score_kernel<<<blocks, 256, 0, stream>>>(h, ei, W1, b1, W2, b2, out, nWaves);
}
